// SparseAttention_35656818492149
// MI455X (gfx1250) — matrix operand order repaired and checked by reading, not compiled, not run on hardware
//
#include <hip/hip_runtime.h>
#include <hip/hip_bf16.h>

// ---------------------------------------------------------------------------
// MI455X (gfx1250) sparse local-attention block, bf16 WMMA pipeline.
//  K0a: x f32 -> bf16
//  K0b: qkv_w / proj_w f32 -> bf16, transposed to W^T [N][K]
//  K1 : QKV GEMM  (x[65536x384] @ w[384x1152] + b) -> bf16 qkv buffer
//       (double-buffered: A via global_load_async_to_lds_b128,
//        B via TDM tensor_load_to_lds)
//  K2 : banded flash attention; softmax row-sums computed by an all-ones
//       WMMA B-tile (no cross-lane shuffles), no running max (scores are
//       provably << exp-overflow range for this problem).
//  K3 : proj GEMM (y[65536x384] @ w[384x384] + b) -> f32 output
// All matrix math: v_wmma_f32_16x16x32_bf16 (wave32).
// ---------------------------------------------------------------------------

typedef __attribute__((ext_vector_type(16))) __bf16 v16bf;
typedef __attribute__((ext_vector_type(8)))  __bf16 v8bf;
typedef __attribute__((ext_vector_type(8)))  float  v8f;
typedef __attribute__((ext_vector_type(4)))  unsigned su4;
typedef __attribute__((ext_vector_type(8)))  unsigned su8;

#define N_ROWS   65536   // B*T
#define CDIM     384
#define NQKV     1152
#define TSEQ     256
#define MMEM     64
#define HDIM     48
#define WINDOW_  64

__device__ __forceinline__ v16bf load_frag2(const __bf16* p0, const __bf16* p1) {
    union { v16bf v; v8bf h[2]; } u;
    u.h[0] = *(const v8bf*)p0;
    u.h[1] = *(const v8bf*)p1;
    return u.v;
}

// ---------------------------------------------------------------------------
__global__ void f32_to_bf16_kernel(const float* __restrict__ src,
                                   __bf16* __restrict__ dst, int n) {
    int i = blockIdx.x * blockDim.x + threadIdx.x;
    if (i < n) dst[i] = (__bf16)src[i];
}

// src[K][N] f32 -> dst[N][K] bf16  (coalesced writes)
__global__ void transpose_f32_to_bf16(const float* __restrict__ src,
                                      __bf16* __restrict__ dst,
                                      int K, int N) {
    int idx = blockIdx.x * blockDim.x + threadIdx.x;
    if (idx < K * N) {
        int n = idx / K;
        int k = idx - n * K;
        dst[idx] = (__bf16)src[(size_t)k * N + n];
    }
}

// ---------------------------------------------------------------------------
// Stage one K-step of tiles into LDS:
//   A tile 64 rows x 32 k   : one async b128 per thread (ASYNCcnt)
//   B tile 128 cols x 32 k  : one TDM tensor_load_to_lds by wave 0 (TENSORcnt)
//     D#: 2B elements, tile_dim0=32 (K), tile_dim1=128 (cols of W^T),
//         tensor_dim0_stride=K; pad 16B after every 64B -> LDS row stride 80B
//         == ldsB stride of 40 bf16 elements.
// ---------------------------------------------------------------------------
__device__ __forceinline__ void stage_tiles(const __bf16* __restrict__ A,
                                            const __bf16* __restrict__ Wt,
                                            __bf16* ldsA, __bf16* ldsB,
                                            int row0, int col0, int k0,
                                            int lda, int K, int tid, int wv) {
    {
        const int arow = tid >> 2;          // 0..63
        const int acol = (tid & 3) * 8;     // 0,8,16,24
        const __bf16* src = A + (size_t)(row0 + arow) * lda + k0 + acol;
        unsigned dst = (unsigned)(size_t)(ldsA + arow * 40 + acol);
        unsigned long long ga = (unsigned long long)(size_t)src;
        asm volatile("global_load_async_to_lds_b128 %0, %1, off"
                     :: "v"(dst), "v"(ga) : "memory");
    }
    if (wv == 0) {
        unsigned long long ga =
            (unsigned long long)(size_t)(Wt + (size_t)col0 * K + k0);
        unsigned ldsb = (unsigned)(size_t)ldsB;
        su4 g0;
        g0[0] = 1u;                                            // count=1
        g0[1] = ldsb;                                          // lds_addr
        g0[2] = (unsigned)ga;                                  // global_addr lo
        g0[3] = (unsigned)((ga >> 32) & 0x01ffffffull)
              | (2u << 30);                                    // addr hi | type=2
        su8 g1;
        g1[0] = (1u << 16)        // data_size = 2 bytes
              | (1u << 20)        // pad_enable
              | (3u << 22)        // pad_interval: every 8x8B = 64B
              | (3u << 25);       // pad_amount: 4 DWORDs = 16B
        g1[1] = ((unsigned)K & 0xffffu) << 16;                 // tensor_dim0 lo16
        g1[2] = (((unsigned)K >> 16) & 0xffffu)                // tensor_dim0 hi16
              | (1152u << 16);                                 // tensor_dim1 lo16
        g1[3] = (32u << 16);                                   // tile_dim0 = 32
        g1[4] = 128u;                                          // tile_dim1 = 128
        g1[5] = (unsigned)K;                                   // dim0_stride lo32
        g1[6] = 0u;
        g1[7] = 0u;
        asm volatile("tensor_load_to_lds %0, %1" :: "s"(g0), "s"(g1) : "memory");
    }
}

// ---------------------------------------------------------------------------
// Tiled GEMM: Out[row, col] = A[row, :K] * Wt[col, :K] + bias[col]
// Block: 256 threads (8 waves). Tile 64 rows x 128 cols, double-buffered K=32.
// Wave w: rows (w&3)*16..+15, cols (w>>2)*64..+63 (4 WMMA tiles).
// ---------------------------------------------------------------------------
template <int OUT_F32>
__global__ __launch_bounds__(256)
void gemm_bf16_wmma(const __bf16* __restrict__ A,
                    const __bf16* __restrict__ Wt,   // [N][K]
                    const float* __restrict__ bias,
                    void* __restrict__ Out,
                    int lda, int ldo, int K) {
    __shared__ __bf16 ldsA[2][64 * 40];     // [row][k], stride 40
    __shared__ __bf16 ldsB[2][128 * 40];    // [n][k],  stride 40 (TDM-padded)

    const int tid  = threadIdx.x;
    const int lane = tid & 31;
    const int wv   = tid >> 5;
    const int hi   = lane >> 4;
    const int ln   = lane & 15;

    const int row0 = blockIdx.y * 64;
    const int col0 = blockIdx.x * 128;
    const int rw   = wv & 3;
    const int cb   = (wv >> 2) * 64;

    v8f acc[4];
#pragma unroll
    for (int t = 0; t < 4; ++t)
#pragma unroll
        for (int j = 0; j < 8; ++j) acc[t][j] = 0.0f;

    const int NIT = K >> 5;

    stage_tiles(A, Wt, ldsA[0], ldsB[0], row0, col0, 0, lda, K, tid, wv);

    for (int i = 0; i < NIT; ++i) {
        const int cur = i & 1;
        const bool more = (i + 1) < NIT;
        if (more)
            stage_tiles(A, Wt, ldsA[1 - cur], ldsB[1 - cur],
                        row0, col0, (i + 1) * 32, lda, K, tid, wv);

        if (more) asm volatile("s_wait_asynccnt 0x1" ::: "memory");
        else      asm volatile("s_wait_asynccnt 0x0" ::: "memory");
        if (wv == 0) {
            if (more) __builtin_amdgcn_s_wait_tensorcnt(1);
            else      __builtin_amdgcn_s_wait_tensorcnt(0);
        }
        __syncthreads();

        // A fragment (16x32): lane<16 K {0..15}; lane>=16 {16..31}
        const __bf16* ab = &ldsA[cur][(rw * 16 + ln) * 40];
        v16bf a = load_frag2(ab + hi * 16, ab + hi * 16 + 8);
#pragma unroll
        for (int t = 0; t < 4; ++t) {
            // B fragment (32x16): lane n holds 16 contiguous K of column n
            const __bf16* bb = &ldsB[cur][(cb + t * 16 + ln) * 40 + hi * 16];
            v16bf b = load_frag2(bb, bb + 8);
            acc[t] = __builtin_amdgcn_wmma_f32_16x16x32_bf16(
                false, a, false, b, (short)0, acc[t], false, false);
        }
        asm volatile("s_wait_dscnt 0x0" ::: "memory");
        __syncthreads();
    }

    // epilogue: bias + store. C layout: M = j + 8*hi, N = ln
    const int mbase = row0 + rw * 16 + 8 * hi;
#pragma unroll
    for (int t = 0; t < 4; ++t) {
        const int col = col0 + cb + t * 16 + ln;
        const float bv = bias[col];
#pragma unroll
        for (int j = 0; j < 8; ++j) {
            const float r = acc[t][j] + bv;
            const size_t off = (size_t)(mbase + j) * ldo + col;
            if (OUT_F32) ((float*)Out)[off] = r;
            else         ((__bf16*)Out)[off] = (__bf16)r;
        }
    }
}

// ---------------------------------------------------------------------------
// Banded attention. Grid: B*H*2 blocks; block handles 128 queries (8 waves,
// one 16-query tile per wave). Key union per block: 64 mem + 192 band keys.
// WMMA for QK^T and PV; row sums of exp(S) via an all-ones WMMA B-tile
// (result lands in every lane, elementwise-aligned with the y accumulators).
// No running max: scores here are O(1) << 88, so exp cannot overflow, and
// softmax without max-subtraction is mathematically identical.
// ---------------------------------------------------------------------------
__global__ __launch_bounds__(256)
void attn_banded_wmma(const __bf16* __restrict__ qkv,   // [N_ROWS][1152]
                      const float* __restrict__ mem,    // [64][384]
                      __bf16* __restrict__ ybuf) {      // [N_ROWS][384]
    __shared__ __bf16 ldsK[256 * 56];    // [key][d], d 0..47 data, 48..55 zero
    __shared__ __bf16 ldsVt[48 * 256];   // transposed: [d][key]
    __shared__ __bf16 ldsP[8 * 16 * 32]; // per-wave P tile [m][32 keys]
    __shared__ __bf16 ldsZ[16];          // zero region for padded K chunk
    __shared__ __bf16 ldsOne[16];        // ones for the row-sum B-tile

    const int tid  = threadIdx.x;
    const int lane = tid & 31;
    const int wv   = tid >> 5;
    const int hi   = lane >> 4;
    const int ln   = lane & 15;

    const int half = blockIdx.x & 1;
    const int bh   = blockIdx.x >> 1;
    const int b    = bh >> 3;            // / HEADS
    const int h    = bh & 7;
    const int q0b  = half * 128;
    const int j0   = q0b - 64;           // self key at slot 64 has index j0

    const float scale = rsqrtf((float)HDIM);

    // ---- stage K/V: one key per thread ----
    {
        const int kk = tid;
        __bf16* krow = &ldsK[kk * 56];
        if (kk < MMEM) {
            const float* ms = mem + (size_t)kk * CDIM + h * HDIM;
#pragma unroll
            for (int d = 0; d < HDIM; ++d) {
                __bf16 kv = (__bf16)ms[d];
                krow[d] = kv;
                ldsVt[d * 256 + kk] = kv;     // mem acts as both K and V
            }
        } else {
            const int j = j0 + kk - MMEM;
            if (j >= 0 && j < TSEQ) {
                const __bf16* ks = qkv + (size_t)(b * TSEQ + j) * NQKV
                                       + CDIM + h * HDIM;
                const __bf16* vs = ks + CDIM;
#pragma unroll
                for (int e = 0; e < 6; ++e)
                    ((v8bf*)krow)[e] = ((const v8bf*)ks)[e];
#pragma unroll
                for (int d = 0; d < HDIM; ++d)
                    ldsVt[d * 256 + kk] = vs[d];
            } else {
#pragma unroll
                for (int d = 0; d < HDIM; ++d) {
                    krow[d] = (__bf16)0.0f;
                    ldsVt[d * 256 + kk] = (__bf16)0.0f;
                }
            }
        }
#pragma unroll
        for (int d = HDIM; d < 56; ++d) krow[d] = (__bf16)0.0f;
        if (tid < 16) ldsZ[tid]   = (__bf16)0.0f;
        if (tid < 16) ldsOne[tid] = (__bf16)1.0f;
    }
    __syncthreads();

    const int qbase = q0b + wv * 16;
    const int rowq  = b * TSEQ + qbase;

    // ---- Q fragments for K-chunks c=0 (d 0..31) and c=1 (d 32..63, padded) --
    v16bf qf[2];
    {
        const __bf16* qs = qkv + (size_t)(rowq + ln) * NQKV + h * HDIM;
        qf[0] = load_frag2(qs + hi * 16, qs + hi * 16 + 8);
        union { v16bf v; v8bf hh[2]; } u;
        v8bf z8;
#pragma unroll
        for (int e = 0; e < 8; ++e) z8[e] = (__bf16)0.0f;
        u.hh[0] = hi ? z8 : *(const v8bf*)(qs + 32);
        u.hh[1] = hi ? z8 : *(const v8bf*)(qs + 40);   // d 48..63 -> zero
        qf[1] = u.v;
    }

    // all-ones B fragment (broadcast read: all lanes same 32B)
    const v16bf bones = load_frag2(ldsOne, ldsOne + 8);

    // yacc[0..2] = sum exp(S)*V tiles; yacc[3] = sum exp(S)  (row sums)
    v8f yacc[4];
#pragma unroll
    for (int t = 0; t < 4; ++t)
#pragma unroll
        for (int j = 0; j < 8; ++j) yacc[t][j] = 0.0f;

    __bf16* myP = &ldsP[wv * 512];

    // key-pair range needed by this wave (mem pairs 0,1 always; band window)
    const int plo = (64 + wv * 16) >> 5;
    const int phi = (143 + wv * 16) >> 5;

    for (int p = 0; p < 8; ++p) {
        if (!(p < 2 || (p >= plo && p <= phi))) continue;   // wave-uniform
        const int kb = p * 32;

        // ---- S = Q K^T for two 16-key tiles ----
        v8f s[2];
#pragma unroll
        for (int t = 0; t < 2; ++t)
#pragma unroll
            for (int j = 0; j < 8; ++j) s[t][j] = 0.0f;
#pragma unroll
        for (int t = 0; t < 2; ++t) {
            const int keyn = kb + t * 16 + ln;
#pragma unroll
            for (int c = 0; c < 2; ++c) {
                const __bf16* bp = (c == 1 && hi)
                    ? &ldsZ[0]
                    : &ldsK[keyn * 56 + c * 32 + hi * 16];
                v16bf bk = load_frag2(bp, bp + 8);
                s[t] = __builtin_amdgcn_wmma_f32_16x16x32_bf16(
                    false, qf[c], false, bk, (short)0, s[t], false, false);
            }
        }

        // ---- mask, exp, write P tile (no cross-lane ops) ----
        const int key0 = kb + ln;
        const int key1 = kb + 16 + ln;
#pragma unroll
        for (int j = 0; j < 8; ++j) {
            const int iq = qbase + j + 8 * hi;
            float sv0 = s[0][j] * scale;
            float sv1 = s[1][j] * scale;
            bool v0 = key0 < MMEM;
            if (!v0) { int jj = j0 + key0 - MMEM;
                       v0 = (jj >= 0) & (jj <= iq) & (iq - jj <= WINDOW_); }
            bool v1 = key1 < MMEM;
            if (!v1) { int jj = j0 + key1 - MMEM;
                       v1 = (jj >= 0) & (jj <= iq) & (iq - jj <= WINDOW_); }
            const float e0 = v0 ? __expf(sv0) : 0.0f;
            const float e1 = v1 ? __expf(sv1) : 0.0f;
            const int m = j + 8 * hi;
            myP[m * 32 + ln]      = (__bf16)e0;
            myP[m * 32 + 16 + ln] = (__bf16)e1;
        }
        asm volatile("s_wait_dscnt 0x0" ::: "memory");   // wave-local LDS RAW

        // ---- y += P V ; rowsum += P * ones ----
        {
            const __bf16* pr = &myP[ln * 32];
            v16bf pa = load_frag2(pr + hi * 16, pr + hi * 16 + 8);
#pragma unroll
            for (int t = 0; t < 3; ++t) {
                const __bf16* vb = &ldsVt[(t * 16 + ln) * 256 + kb + hi * 16];
                v16bf bv = load_frag2(vb, vb + 8);
                yacc[t] = __builtin_amdgcn_wmma_f32_16x16x32_bf16(
                    false, pa, false, bv, (short)0, yacc[t], false, false);
            }
            yacc[3] = __builtin_amdgcn_wmma_f32_16x16x32_bf16(
                false, pa, false, bones, (short)0, yacc[3], false, false);
        }
    }

    // ---- finalize: divide by row sums, store bf16 y ----
#pragma unroll
    for (int t = 0; t < 3; ++t) {
#pragma unroll
        for (int j = 0; j < 8; ++j) {
            const int m = j + 8 * hi;
            const size_t off = (size_t)(rowq + m) * CDIM + h * HDIM + t * 16 + ln;
            ybuf[off] = (__bf16)(yacc[t][j] / yacc[3][j]);
        }
    }
}

// ---------------------------------------------------------------------------
extern "C" void kernel_launch(void* const* d_in, const int* in_sizes, int n_in,
                              void* d_out, int out_size, void* d_ws, size_t ws_size,
                              hipStream_t stream) {
    const float* x      = (const float*)d_in[0];
    const float* mem    = (const float*)d_in[1];
    const float* qkv_w  = (const float*)d_in[2];
    const float* qkv_b  = (const float*)d_in[3];
    const float* proj_w = (const float*)d_in[4];
    const float* proj_b = (const float*)d_in[5];
    float* out = (float*)d_out;

    // workspace layout (bf16):
    //   [0)          x bf16        25165824 el ->  50331648 B
    //   [50331648)   qkv_w^T bf16    442368 el ->    884736 B
    //   [51216384)   proj_w^T bf16   147456 el ->    294912 B
    //   [51511296)   qkv buffer  65536*1152   -> 150994944 B
    //   [202506240)  y buffer    65536*384    ->  50331648 B   (~253 MB total)
    char* ws = (char*)d_ws;
    __bf16* xbf    = (__bf16*)(ws);
    __bf16* wqkvT  = (__bf16*)(ws + 50331648ull);
    __bf16* wprojT = (__bf16*)(ws + 51216384ull);
    __bf16* qkvbuf = (__bf16*)(ws + 51511296ull);
    __bf16* ybuf   = (__bf16*)(ws + 202506240ull);

    f32_to_bf16_kernel<<<(25165824 + 255) / 256, 256, 0, stream>>>(x, xbf, 25165824);
    transpose_f32_to_bf16<<<(442368 + 255) / 256, 256, 0, stream>>>(qkv_w, wqkvT,
                                                                    CDIM, NQKV);
    transpose_f32_to_bf16<<<(147456 + 255) / 256, 256, 0, stream>>>(proj_w, wprojT,
                                                                    CDIM, CDIM);

    // QKV GEMM: 65536 x 1152, K=384, out = bf16 qkv buffer
    dim3 g1(NQKV / 128, N_ROWS / 64);
    gemm_bf16_wmma<0><<<g1, 256, 0, stream>>>(xbf, wqkvT, qkv_b, qkvbuf,
                                              CDIM, NQKV, CDIM);

    // attention: B*H*2 blocks of 256 threads
    attn_banded_wmma<<<256 * 8 * 2, 256, 0, stream>>>(qkvbuf, mem, ybuf);

    // projection GEMM: 65536 x 384, K=384, out = f32
    dim3 g2(CDIM / 128, N_ROWS / 64);
    gemm_bf16_wmma<1><<<g2, 256, 0, stream>>>(ybuf, wprojT, proj_b, out,
                                              CDIM, CDIM, CDIM);
}
